// PointNetSetAbstraction_25185688223746
// MI455X (gfx1250) — matrix operand order repaired and checked by reading, not compiled, not run on hardware
//
#include <hip/hip_runtime.h>
#include <hip/hip_bf16.h>

typedef __attribute__((ext_vector_type(16))) _Float16 v16h;
typedef __attribute__((ext_vector_type(8)))  _Float16 v8h;
typedef __attribute__((ext_vector_type(8)))  float    v8f;

#define BB   16
#define NN   4096
#define SS   1024
#define KK   32
#define RAD2 0.04f
#define EPSV 1e-5f
#define MTOT ((long)BB * SS * KK)   // 524288 grouped points

// ---------------------------------------------------------------- FPS ------
// One 256-thread block per batch. Each thread keeps its 16 points + running
// min-distance in registers; LDS tree argmax per iteration (1024 iterations).
__global__ void __launch_bounds__(256)
fps_kernel(const float* __restrict__ xyz, int* __restrict__ cent) {
  const int b = blockIdx.x;
  const int t = threadIdx.x;
  const float* base = xyz + (long)b * 3 * NN;
  float px[16], py[16], pz[16], dist[16];
#pragma unroll
  for (int j = 0; j < 16; ++j) {
    int i = t + 256 * j;
    px[j] = base[i];
    py[j] = base[NN + i];
    pz[j] = base[2 * NN + i];
    dist[j] = 1e10f;
  }
  __shared__ float redv[256];
  __shared__ int   redi[256];
  __shared__ int   sfar;
  if (t == 0) sfar = 0;
  __syncthreads();
  for (int s = 0; s < SS; ++s) {
    int far = sfar;
    if (t == 0) cent[b * SS + s] = far;
    float cx = base[far], cy = base[NN + far], cz = base[2 * NN + far];
    float best = -1.0f; int bi = 0x7fffffff;
#pragma unroll
    for (int j = 0; j < 16; ++j) {
      float dx = px[j] - cx, dy = py[j] - cy, dz = pz[j] - cz;
      float d  = dx * dx + dy * dy + dz * dz;
      float dm = fminf(dist[j], d);
      dist[j] = dm;
      int i = t + 256 * j;
      if (dm > best || (dm == best && i < bi)) { best = dm; bi = i; }
    }
    redv[t] = best; redi[t] = bi;
    __syncthreads();
    for (int off = 128; off > 0; off >>= 1) {
      if (t < off) {
        float ov = redv[t + off]; int oi = redi[t + off];
        if (ov > redv[t] || (ov == redv[t] && oi < redi[t])) {
          redv[t] = ov; redi[t] = oi;
        }
      }
      __syncthreads();
    }
    if (t == 0) sfar = redi[0];
    __syncthreads();
  }
}

// ------------------------------------------------------- gather new_xyz ----
__global__ void gather_newxyz(const float* __restrict__ xyz,
                              const int* __restrict__ cent,
                              float* __restrict__ out) {
  int i = blockIdx.x * blockDim.x + threadIdx.x;
  if (i >= BB * 3 * SS) return;
  int s = i % SS;
  int c = (i / SS) % 3;
  int b = i / (3 * SS);
  out[i] = xyz[(long)b * 3 * NN + c * NN + cent[b * SS + s]];
}

// ------------------------------------------------------------ ball query ---
// One thread per centroid; ascending scan reproduces sort()[:K] semantics.
__global__ void ball_query_kernel(const float* __restrict__ xyz,
                                  const float* __restrict__ new_xyz, // (B,3,S)
                                  int* __restrict__ gidx) {
  int tid = blockIdx.x * blockDim.x + threadIdx.x;
  if (tid >= BB * SS) return;
  int b = tid / SS, s = tid - b * SS;
  const float* base = xyz + (long)b * 3 * NN;
  const float* nb   = new_xyz + (long)b * 3 * SS;
  float cx = nb[s], cy = nb[SS + s], cz = nb[2 * SS + s];
  int* dst = gidx + (long)tid * KK;
  int cnt = 0, first = 0;
  for (int n = 0; n < NN; ++n) {
    float dx = base[n] - cx, dy = base[NN + n] - cy, dz = base[2 * NN + n] - cz;
    float d = dx * dx + dy * dy + dz * dz;
    if (d <= RAD2) {
      if (cnt == 0) first = n;
      dst[cnt] = n;
      if (++cnt == KK) break;
    }
  }
  for (int k = cnt; k < KK; ++k) dst[k] = first;
}

// -------------------------------------------------------------- grouping ---
// Build x0 as f16 [m][32]: ch 0-2 relative xyz, ch 3-5 points, ch 6-31 zero
// (zero-pad so layer0 can use the 16x16x32 f16 WMMA directly).
__global__ void group_kernel(const float* __restrict__ xyz,
                             const float* __restrict__ pts,
                             const float* __restrict__ new_xyz,
                             const int* __restrict__ gidx,
                             _Float16* __restrict__ x0) {
  long tid = (long)blockIdx.x * blockDim.x + threadIdx.x;
  if (tid >= MTOT) return;
  long bs = tid / KK;
  int s = (int)(bs % SS);
  int b = (int)(bs / SS);
  int idx = gidx[tid];
  if (idx >= NN) idx = NN - 1;        // JAX gather clamp semantics
  _Float16* row = x0 + tid * 32;
  const float* xb = xyz + (long)b * 3 * NN;
  const float* pb = pts + (long)b * 3 * NN;
  const float* nb = new_xyz + (long)b * 3 * SS;
#pragma unroll
  for (int c = 0; c < 3; ++c) row[c]     = (_Float16)(xb[c * NN + idx] - nb[c * SS + s]);
#pragma unroll
  for (int c = 0; c < 3; ++c) row[3 + c] = (_Float16)pb[c * NN + idx];
#pragma unroll
  for (int c = 6; c < 32; ++c) row[c] = (_Float16)0.f;
}

// ------------------------------------------------------ weight f32 -> f16 --
__global__ void wconv_kernel(const float* __restrict__ w, _Float16* __restrict__ wh,
                             int O, int Cs, int Cd) {
  int i = blockIdx.x * blockDim.x + threadIdx.x;
  if (i >= O * Cd) return;
  int c = i % Cd, o = i / Cd;
  wh[i] = (c < Cs) ? (_Float16)w[o * Cs + c] : (_Float16)0.f;
}

__global__ void zero_kernel(float* p, int n) {
  int i = blockIdx.x * blockDim.x + threadIdx.x;
  if (i < n) p[i] = 0.f;
}

// -------------------------------------------------------- WMMA fragments ---
// A (weights, row-major O x C): 16x32 f16 tile.  Lane L (half=L/16, row=L%16)
// holds K pairs {0..7} and {16..23} offset by half*8  -> two contiguous v8h.
__device__ __forceinline__ v16h load_frag_a(const _Float16* __restrict__ w, int C,
                                            int o0, int c0, int lane) {
  const int half_ = lane >> 4;
  const int row   = lane & 15;
  const _Float16* p = w + (o0 + row) * C + c0 + half_ * 8;
  v8h lo = *(const v8h*)(p);        // K = base+0..7
  v8h hi = *(const v8h*)(p + 16);   // K = base+16..23
  v16h a;
#pragma unroll
  for (int j = 0; j < 8; ++j) { a[j] = lo[j]; a[8 + j] = hi[j]; }
  return a;
}

// B (activations, row-major M x C): 32x16 f16 tile.  Lane L = column L%16,
// K pairs {0..7} and {16..23} offset by (L/16)*8  -> two contiguous v8h.
__device__ __forceinline__ v16h load_frag_b(const _Float16* __restrict__ x, int C,
                                            long m0, int c0, int lane) {
  const int half_ = lane >> 4;
  const int col   = lane & 15;
  const _Float16* p = x + (m0 + col) * C + c0 + half_ * 8;
  v8h lo = *(const v8h*)(p);
  v8h hi = *(const v8h*)(p + 16);
  v16h b;
#pragma unroll
  for (int j = 0; j < 8; ++j) { b[j] = lo[j]; b[8 + j] = hi[j]; }
  return b;
}

// ------------------------------------------------------------ WMMA GEMM ----
// y[m][o] = sum_c x[m][c] * w[o][c] + bias[o], f32 accum, stored f16 in the
// next layer's [m][c] input layout. 8 waves/block, 16 points per wave,
// B-fragments reused across all output-channel tiles.
template <int CIN, int OCH>
__global__ void __launch_bounds__(256)
gemm_wmma(const _Float16* __restrict__ x, const _Float16* __restrict__ w,
          const float* __restrict__ bias, _Float16* __restrict__ y) {
  const int lane  = threadIdx.x & 31;
  const int wave  = threadIdx.x >> 5;
  const long m0   = ((long)blockIdx.x * 8 + wave) * 16;
  const int half_ = lane >> 4;
  const int col   = lane & 15;

  constexpr int KSTEPS = CIN / 32;
  v16h bfrag[KSTEPS];
#pragma unroll
  for (int ks = 0; ks < KSTEPS; ++ks)
    bfrag[ks] = load_frag_b(x, CIN, m0, ks * 32, lane);

#pragma unroll
  for (int ot = 0; ot < OCH / 16; ++ot) {
    const int o0 = ot * 16;
    v8f acc;
#pragma unroll
    for (int r = 0; r < 8; ++r) acc[r] = bias[o0 + half_ * 8 + r];
#pragma unroll
    for (int ks = 0; ks < KSTEPS; ++ks) {
      v16h afrag = load_frag_a(w, CIN, o0, ks * 32, lane);
      acc = __builtin_amdgcn_wmma_f32_16x16x32_f16(false, afrag, false, bfrag[ks],
                                                   (short)0, acc, false, false);
    }
    // D layout: VGPR r -> out-channel o0 + half*8 + r, column = point m0+col
    _Float16* dst = y + (m0 + col) * OCH + o0 + half_ * 8;
#pragma unroll
    for (int r = 0; r < 8; ++r) dst[r] = (_Float16)acc[r];
  }
}

// --------------------------------------------------------- BN statistics ---
template <int OCH>
__global__ void __launch_bounds__(256)
bn_stats(const _Float16* __restrict__ y, float* __restrict__ stats) {
  constexpr int ROWS = 256 / OCH;
  const int c = threadIdx.x % OCH;
  const int r = threadIdx.x / OCH;
  float s = 0.f, s2 = 0.f;
  for (long m = (long)blockIdx.x * ROWS + r; m < MTOT; m += (long)gridDim.x * ROWS) {
    float v = (float)y[m * OCH + c];
    s += v; s2 += v * v;
  }
  __shared__ float ls[256], ls2[256];
  ls[threadIdx.x] = s; ls2[threadIdx.x] = s2;
  __syncthreads();
  if (r == 0) {
#pragma unroll
    for (int k = 1; k < ROWS; ++k) { s += ls[c + k * OCH]; s2 += ls2[c + k * OCH]; }
    atomicAdd(&stats[c], s);
    atomicAdd(&stats[OCH + c], s2);
  }
}

// ------------------------------------------------- BN + ReLU (to f16 in) ---
template <int OCH>
__global__ void bn_relu(const _Float16* __restrict__ y, const float* __restrict__ stats,
                        const float* __restrict__ g, const float* __restrict__ be,
                        _Float16* __restrict__ xo) {
  long i = (long)blockIdx.x * blockDim.x + threadIdx.x;
  if (i >= MTOT * OCH) return;
  int c = (int)(i % OCH);
  const float inv_m = 1.0f / (float)MTOT;
  float mean = stats[c] * inv_m;
  float var  = stats[OCH + c] * inv_m - mean * mean;
  float v = ((float)y[i] - mean) * rsqrtf(var + EPSV) * g[c] + be[c];
  xo[i] = (_Float16)fmaxf(v, 0.f);
}

// ------------------------------------ layer-2 BN + ReLU fused with max_K ---
__global__ void bn_relu_max(const _Float16* __restrict__ y, const float* __restrict__ stats,
                            const float* __restrict__ g, const float* __restrict__ be,
                            float* __restrict__ out) {
  long i = (long)blockIdx.x * blockDim.x + threadIdx.x;     // index over (b,o,s)
  const long total = (long)BB * 128 * SS;
  if (i >= total) return;
  int s = (int)(i % SS);
  int o = (int)((i / SS) % 128);
  int b = (int)(i / ((long)128 * SS));
  const float inv_m = 1.0f / (float)MTOT;
  float mean  = stats[o] * inv_m;
  float var   = stats[128 + o] * inv_m - mean * mean;
  float scale = rsqrtf(var + EPSV) * g[o];
  float shift = be[o] - mean * scale;
  const _Float16* p = y + ((long)(b * SS + s) * KK) * 128 + o;
  float mx = 0.f;                       // ReLU output is >= 0
#pragma unroll 8
  for (int k = 0; k < KK; ++k) {
    float v = (float)p[(long)k * 128] * scale + shift;
    mx = fmaxf(mx, fmaxf(v, 0.f));
  }
  out[i] = mx;
}

// ------------------------------------------------------------------ host ---
extern "C" void kernel_launch(void* const* d_in, const int* in_sizes, int n_in,
                              void* d_out, int out_size, void* d_ws, size_t ws_size,
                              hipStream_t stream) {
  (void)in_sizes; (void)n_in; (void)out_size; (void)ws_size;
  const float* xyz = (const float*)d_in[0];
  const float* pts = (const float*)d_in[1];
  const float* w0  = (const float*)d_in[2];
  const float* b0  = (const float*)d_in[3];
  const float* g0  = (const float*)d_in[4];
  const float* be0 = (const float*)d_in[5];
  const float* w1  = (const float*)d_in[6];
  const float* b1  = (const float*)d_in[7];
  const float* g1  = (const float*)d_in[8];
  const float* be1 = (const float*)d_in[9];
  const float* w2  = (const float*)d_in[10];
  const float* b2  = (const float*)d_in[11];
  const float* g2  = (const float*)d_in[12];
  const float* be2 = (const float*)d_in[13];

  float* out_xyz = (float*)d_out;                  // (B,3,S)
  float* out_np  = out_xyz + (long)BB * 3 * SS;    // (B,128,S)

  char* ws = (char*)d_ws;
  int*      cent  = (int*)(ws + 0);                         // 64 KB
  int*      gidx  = (int*)(ws + (1L  << 20));               // 2 MB
  _Float16* wh0   = (_Float16*)(ws + (4L  << 20));          // 64x32 f16
  _Float16* wh1   = (_Float16*)(ws + (4L  << 20) + 65536);  // 64x64 f16
  _Float16* wh2   = (_Float16*)(ws + (4L  << 20) + 131072); // 128x64 f16
  float*    stats = (float*)(ws + (4L  << 20) + 262144);    // 256 f32
  _Float16* x0    = (_Float16*)(ws + (8L   << 20));         // M*32 f16 = 32 MB
  _Float16* x1    = (_Float16*)(ws + (48L  << 20));         // M*64 f16 = 64 MB
  _Float16* x2    = (_Float16*)(ws + (120L << 20));         // M*64 f16 = 64 MB
  _Float16* ybuf  = (_Float16*)(ws + (192L << 20));         // M*128 f16 = 128 MB

  // Sampling / grouping
  fps_kernel<<<BB, 256, 0, stream>>>(xyz, cent);
  gather_newxyz<<<(BB * 3 * SS) / 256, 256, 0, stream>>>(xyz, cent, out_xyz);
  ball_query_kernel<<<(BB * SS) / 256, 256, 0, stream>>>(xyz, out_xyz, gidx);
  group_kernel<<<(int)(MTOT / 256), 256, 0, stream>>>(xyz, pts, out_xyz, gidx, x0);

  // Weight conversion (f32 -> f16, layer0 padded 6 -> 32)
  wconv_kernel<<<8,  256, 0, stream>>>(w0, wh0, 64, 6, 32);
  wconv_kernel<<<16, 256, 0, stream>>>(w1, wh1, 64, 64, 64);
  wconv_kernel<<<32, 256, 0, stream>>>(w2, wh2, 128, 64, 64);

  const int gemm_blocks = (int)(MTOT / 128);   // 8 waves x 16 points / block
  const int elem_blocks = (int)(MTOT * 64 / 256);

  // Layer 0: 32(pad of 6) -> 64
  gemm_wmma<32, 64><<<gemm_blocks, 256, 0, stream>>>(x0, wh0, b0, ybuf);
  zero_kernel<<<1, 256, 0, stream>>>(stats, 256);
  bn_stats<64><<<1024, 256, 0, stream>>>(ybuf, stats);
  bn_relu<64><<<elem_blocks, 256, 0, stream>>>(ybuf, stats, g0, be0, x1);

  // Layer 1: 64 -> 64
  gemm_wmma<64, 64><<<gemm_blocks, 256, 0, stream>>>(x1, wh1, b1, ybuf);
  zero_kernel<<<1, 256, 0, stream>>>(stats, 256);
  bn_stats<64><<<1024, 256, 0, stream>>>(ybuf, stats);
  bn_relu<64><<<elem_blocks, 256, 0, stream>>>(ybuf, stats, g1, be1, x2);

  // Layer 2: 64 -> 128, BN+ReLU fused into max over K
  gemm_wmma<64, 128><<<gemm_blocks, 256, 0, stream>>>(x2, wh2, b2, ybuf);
  zero_kernel<<<1, 256, 0, stream>>>(stats, 256);
  bn_stats<128><<<1024, 256, 0, stream>>>(ybuf, stats);
  bn_relu_max<<<(int)((long)BB * 128 * SS / 256), 256, 0, stream>>>(ybuf, stats, g2, be2, out_np);
}
